// MultiTaskGAT_89756226551876
// MI455X (gfx1250) — compile-verified, matrix operand order checked by reading, not run on hardware
//
#include <hip/hip_runtime.h>
#include <hip/hip_bf16.h>

// ---------------------------------------------------------------------------
// Types for WMMA (gfx1250, wave32)
// ---------------------------------------------------------------------------
typedef __bf16    v16bf  __attribute__((ext_vector_type(16)));
typedef float     v8f    __attribute__((ext_vector_type(8)));
typedef unsigned  u32x4  __attribute__((ext_vector_type(4)));
typedef float     f32x4  __attribute__((ext_vector_type(4)));
typedef float     f32x2  __attribute__((ext_vector_type(2)));

union Frag16 { u32x4 u[2]; v16bf v; };
union Acc8   { v8f v; float f[8]; };

__device__ __forceinline__ unsigned short f2bf(float x) {
  union { __bf16 b; unsigned short s; } cv;
  cv.b = (__bf16)x;
  return cv.s;
}

// monotonic uint mapping for float atomic-max
__device__ __forceinline__ unsigned fmapf(float f) {
  unsigned u = __float_as_uint(f);
  return (u & 0x80000000u) ? ~u : (u | 0x80000000u);
}
__device__ __forceinline__ float funmapf(unsigned u) {
  return (u & 0x80000000u) ? __uint_as_float(u & 0x7FFFFFFFu)
                           : __uint_as_float(~u);
}

// ---------------------------------------------------------------------------
// bf16 WMMA GEMM:  C[M,Nc] = A[M,K] @ B[K,Nc] (+bias)
// concat_mode==1: A is the virtual [M,2304] concat of
//   nearest_interp(rand,768) | nearest_interp(func,768) | emb
//   (region is uniform per 32-wide K tile -> uniform branch, no per-elem test)
// Block: 256 threads (8 waves). Tile: 256(M) x 64(N), BK=32.
// Each wave owns a 32x64 strip: 8 WMMAs / K-step, 12 b128 LDS frag loads.
// ---------------------------------------------------------------------------
__global__ __launch_bounds__(256)
void gemm_bf16_wmma(const float* __restrict__ A, const float* __restrict__ B,
                    const float* __restrict__ bias, float* __restrict__ C,
                    int M, int K, int Nc,
                    const float* __restrict__ rnd, const float* __restrict__ fnc,
                    const float* __restrict__ emb, int concat_mode)
{
  // padded rows: 40 elems = 80 bytes (multiple of 16B, bank-spread)
  __shared__ __align__(16) unsigned short As[256 * 40];
  __shared__ __align__(16) unsigned short Bt[64 * 40];

  const int tid  = threadIdx.x;
  const int m0   = blockIdx.y * 256;
  const int n0   = blockIdx.x * 64;
  const int wv   = tid >> 5;          // wave id 0..7 -> 32-row M strip
  const int ln   = tid & 31;
  const int lr   = ln & 15;
  const int half = (ln >> 4) * 8;     // K sub-offset per ISA 16-bit layout

  Acc8 acc[2][4];
#pragma unroll
  for (int mi = 0; mi < 2; ++mi)
#pragma unroll
    for (int j = 0; j < 4; ++j)
#pragma unroll
      for (int r = 0; r < 8; ++r) acc[mi][j].f[r] = 0.0f;

  const int tr = tid >> 3;          // 0..31 (A staging row)
  const int tc = (tid & 7) * 4;     // 0..28 (A staging col group)
  const int bk = tid >> 4;          // 0..15 (B staging k)
  const int bn = (tid & 15) * 4;    // 0..60 (B staging col group)

  for (int k0 = 0; k0 < K; k0 += 32) {
    // ---- stage A tile [256 x 32] f32 -> bf16 LDS ----
#pragma unroll
    for (int rp = 0; rp < 8; ++rp) {
      const int row = rp * 32 + tr;
      const int gm  = m0 + row;
      float v0 = 0.f, v1 = 0.f, v2 = 0.f, v3 = 0.f;
      if (gm < M) {
        if (!concat_mode) {
          const f32x4 f = *(const f32x4*)(A + (size_t)gm * K + k0 + tc);
          v0 = f[0]; v1 = f[1]; v2 = f[2]; v3 = f[3];
        } else if (k0 < 768) {            // nearest-interp of rand (64 -> 768)
          const float* rp_ = rnd + (size_t)gm * 64;
          const int kb = k0 + tc;
          v0 = rp_[(kb + 0) / 12]; v1 = rp_[(kb + 1) / 12];
          v2 = rp_[(kb + 2) / 12]; v3 = rp_[(kb + 3) / 12];
        } else if (k0 < 1536) {           // nearest-interp of func (384 -> 768)
          const f32x2 f = *(const f32x2*)(fnc + (size_t)gm * 384 + ((k0 - 768 + tc) >> 1));
          v0 = f[0]; v1 = f[0]; v2 = f[1]; v3 = f[1];
        } else {                          // emb passthrough
          const f32x4 f = *(const f32x4*)(emb + (size_t)gm * 768 + (k0 - 1536) + tc);
          v0 = f[0]; v1 = f[1]; v2 = f[2]; v3 = f[3];
        }
      }
      unsigned short* p = &As[row * 40 + tc];
      p[0] = f2bf(v0); p[1] = f2bf(v1); p[2] = f2bf(v2); p[3] = f2bf(v3);
    }
    // ---- stage B tile [32 x 64] f32 -> bf16 LDS, transposed to [64][32] ----
#pragma unroll
    for (int kp = 0; kp < 2; ++kp) {
      const int k = kp * 16 + bk;
      const f32x4 f = *(const f32x4*)(B + (size_t)(k0 + k) * Nc + n0 + bn);
      Bt[(bn + 0) * 40 + k] = f2bf(f[0]);
      Bt[(bn + 1) * 40 + k] = f2bf(f[1]);
      Bt[(bn + 2) * 40 + k] = f2bf(f[2]);
      Bt[(bn + 3) * 40 + k] = f2bf(f[3]);
    }
    __syncthreads();

    // ---- fragments (b128 LDS reads matching ISA 16-bit A/B layouts) ----
    Frag16 fa[2];
#pragma unroll
    for (int mi = 0; mi < 2; ++mi) {
      const int arow = wv * 32 + mi * 16 + lr;
      fa[mi].u[0] = *(const u32x4*)&As[arow * 40 + half];
      fa[mi].u[1] = *(const u32x4*)&As[arow * 40 + 16 + half];
    }
#pragma unroll
    for (int j = 0; j < 4; ++j) {
      Frag16 fb;
      const int brow = j * 16 + lr;
      fb.u[0] = *(const u32x4*)&Bt[brow * 40 + half];
      fb.u[1] = *(const u32x4*)&Bt[brow * 40 + 16 + half];
#pragma unroll
      for (int mi = 0; mi < 2; ++mi) {
        acc[mi][j].v = __builtin_amdgcn_wmma_f32_16x16x32_bf16(
            false, fa[mi].v, false, fb.v, (short)0, acc[mi][j].v, false, false);
      }
    }
    __syncthreads();
  }

  // ---- epilogue: C/D layout (VGPR r -> row r + 8*(ln>>4), col ln&15) ----
#pragma unroll
  for (int mi = 0; mi < 2; ++mi) {
#pragma unroll
    for (int j = 0; j < 4; ++j) {
      const int gn = n0 + j * 16 + lr;
      const float bv = bias ? bias[gn] : 0.0f;
#pragma unroll
      for (int r = 0; r < 8; ++r) {
        const int gm = m0 + wv * 32 + mi * 16 + (ln >> 4) * 8 + r;
        if (gm < M) C[(size_t)gm * Nc + gn] = acc[mi][j].f[r] + bv;
      }
    }
  }
}

// ---------------------------------------------------------------------------
// Per-(node,head) attention projections el/er + init of emax/denom.
// One wave32 per (n,h); coalesced strided loads; shuffle reduction.
// ---------------------------------------------------------------------------
__global__ void rowdot(const float* __restrict__ feat, const float* __restrict__ al,
                       const float* __restrict__ ar, float* __restrict__ el,
                       float* __restrict__ er, unsigned* __restrict__ emax,
                       float* __restrict__ denom, int N, int H, int Dh)
{
  const int idx  = (int)(((size_t)blockIdx.x * blockDim.x + threadIdx.x) >> 5);
  const int lane = threadIdx.x & 31;
  if (idx >= N * H) return;
  const int h = idx % H;
  const float* fp = feat + (size_t)idx * Dh;   // feat layout [n,h,d], idx=n*H+h
  float sl = 0.f, sr = 0.f;
  for (int i = lane; i < Dh; i += 32) {
    const float v = fp[i];
    sl += v * al[h * Dh + i];
    sr += v * ar[h * Dh + i];
  }
#pragma unroll
  for (int o = 16; o > 0; o >>= 1) {
    sl += __shfl_down(sl, o, 32);
    sr += __shfl_down(sr, o, 32);
  }
  if (lane == 0) { el[idx] = sl; er[idx] = sr; emax[idx] = 0u; denom[idx] = 0.f; }
}

// ---------------------------------------------------------------------------
// Edge softmax pass 1: segment max over dst (mapped-uint atomicMax)
// ---------------------------------------------------------------------------
__global__ void edge_max(const int* __restrict__ src, const int* __restrict__ dst,
                         const float* __restrict__ el, const float* __restrict__ er,
                         unsigned* __restrict__ emax, int E, int H)
{
  const int g = blockIdx.x * blockDim.x + threadIdx.x;
  if (g >= E * H) return;
  const int e = g / H, h = g - e * H;
  const int s = src[e], d = dst[e];
  float x = el[s * H + h] + er[d * H + h];
  x = x >= 0.f ? x : 0.2f * x;                 // leaky_relu slope 0.2
  atomicMax(&emax[d * H + h], fmapf(x));
}

// Edge softmax pass 2: w = exp(e - max), segment sum of w
__global__ void edge_exp(const int* __restrict__ src, const int* __restrict__ dst,
                         const float* __restrict__ el, const float* __restrict__ er,
                         const unsigned* __restrict__ emax, float* __restrict__ we,
                         float* __restrict__ denom, int E, int H)
{
  const int g = blockIdx.x * blockDim.x + threadIdx.x;
  if (g >= E * H) return;
  const int e = g / H, h = g - e * H;
  const int s = src[e], d = dst[e];
  float x = el[s * H + h] + er[d * H + h];
  x = x >= 0.f ? x : 0.2f * x;
  const float w = expf(x - funmapf(emax[d * H + h]));
  we[g] = w;
  atomicAdd(&denom[d * H + h], w);
}

// Edge pass 3: rst[dst] += feat[src] * (w/denom[dst]); one wave per (edge,head)
__global__ void edge_agg(const int* __restrict__ src, const int* __restrict__ dst,
                         const float* __restrict__ feat, const float* __restrict__ we,
                         const float* __restrict__ denom, float* __restrict__ rst,
                         int E, int H, int Dh)
{
  const int gw   = (int)(((size_t)blockIdx.x * blockDim.x + threadIdx.x) >> 5);
  const int lane = threadIdx.x & 31;
  if (gw >= E * H) return;
  const int e = gw / H, h = gw - e * H;
  const int s = src[e], d = dst[e];
  const float a = we[gw] / denom[d * H + h];
  const float* fp = feat + ((size_t)s * H + h) * Dh;
  float*       rp = rst  + ((size_t)d * H + h) * Dh;
  for (int i = lane; i < Dh; i += 32) atomicAdd(&rp[i], fp[i] * a);
}

__global__ void add_bias_rows(float* __restrict__ x, const float* __restrict__ b,
                              int total, int F)
{
  const int g = blockIdx.x * blockDim.x + threadIdx.x;
  if (g >= total) return;
  x[g] += b[g % F];
}

// ---------------------------------------------------------------------------
// Per-node head: h = rst2+b2; attn=sigmoid(h.naW+nab); hg += h*attn (atomic);
// node_logits = (relu(h@W1+b1)@W2 + b2)/T. One 128-thread block per node.
// ---------------------------------------------------------------------------
__global__ __launch_bounds__(128)
void node_head(const float* __restrict__ rst2, const float* __restrict__ b2,
               const float* __restrict__ naW, const float* __restrict__ nab,
               const float* __restrict__ W1, const float* __restrict__ bb1,
               const float* __restrict__ W2, const float* __restrict__ bb2,
               const float* __restrict__ temp, float* __restrict__ out,
               float* __restrict__ hg)
{
  const int n = blockIdx.x, t = threadIdx.x;
  __shared__ float sh[128], red[128];
  __shared__ float sattn;
  const float hv = rst2[(size_t)n * 128 + t] + b2[t];
  sh[t]  = hv;
  red[t] = hv * naW[t];
  __syncthreads();
  for (int o = 64; o > 0; o >>= 1) { if (t < o) red[t] += red[t + o]; __syncthreads(); }
  if (t == 0) sattn = 1.0f / (1.0f + expf(-(red[0] + nab[0])));
  __syncthreads();
  atomicAdd(&hg[t], hv * sattn);

  float acc = bb1[t];
  for (int d = 0; d < 128; ++d) acc += sh[d] * W1[d * 128 + t];
  const float tv = acc > 0.f ? acc : 0.f;
  const float T = temp[0];
  for (int k = 0; k < 2; ++k) {
    __syncthreads();
    red[t] = tv * W2[t * 2 + k];
    __syncthreads();
    for (int o = 64; o > 0; o >>= 1) { if (t < o) red[t] += red[t + o]; __syncthreads(); }
    if (t == 0) out[(size_t)n * 2 + k] = (red[0] + bb2[k]) / T;
  }
}

__global__ __launch_bounds__(128)
void graph_head(const float* __restrict__ hg, const float* __restrict__ W1,
                const float* __restrict__ b1, const float* __restrict__ W2,
                const float* __restrict__ b2, const float* __restrict__ temp,
                float* __restrict__ out)
{
  const int t = threadIdx.x;
  __shared__ float sh[128], red[128];
  sh[t] = hg[t];
  __syncthreads();
  float acc = b1[t];
  for (int d = 0; d < 128; ++d) acc += sh[d] * W1[d * 128 + t];
  const float tv = acc > 0.f ? acc : 0.f;
  const float T = temp[0];
  for (int k = 0; k < 2; ++k) {
    __syncthreads();
    red[t] = tv * W2[t * 2 + k];
    __syncthreads();
    for (int o = 64; o > 0; o >>= 1) { if (t < o) red[t] += red[t + o]; __syncthreads(); }
    if (t == 0) out[k] = (red[0] + b2[k]) / T;
  }
}

// ---------------------------------------------------------------------------
extern "C" void kernel_launch(void* const* d_in, const int* in_sizes, int n_in,
                              void* d_out, int out_size, void* d_ws, size_t ws_size,
                              hipStream_t stream)
{
  (void)in_sizes; (void)n_in; (void)out_size; (void)ws_size;
  const float* rand_f = (const float*)d_in[0];
  const float* func_f = (const float*)d_in[1];
  const float* emb_f  = (const float*)d_in[2];
  const int*   src    = (const int*)d_in[3];
  const int*   dst    = (const int*)d_in[4];
  const float* proj_W = (const float*)d_in[5];
  const float* proj_b = (const float*)d_in[6];
  const float* W1     = (const float*)d_in[7];
  const float* al1    = (const float*)d_in[8];
  const float* ar1    = (const float*)d_in[9];
  const float* b1     = (const float*)d_in[10];
  const float* W2     = (const float*)d_in[11];
  const float* al2    = (const float*)d_in[12];
  const float* ar2    = (const float*)d_in[13];
  const float* b2     = (const float*)d_in[14];
  const float* naW    = (const float*)d_in[15];
  const float* nab    = (const float*)d_in[16];
  const float* nmW1   = (const float*)d_in[17];
  const float* nmb1   = (const float*)d_in[18];
  const float* nmW2   = (const float*)d_in[19];
  const float* nmb2   = (const float*)d_in[20];
  const float* gmW1   = (const float*)d_in[21];
  const float* gmb1   = (const float*)d_in[22];
  const float* gmW2   = (const float*)d_in[23];
  const float* gmb2   = (const float*)d_in[24];
  const float* temp   = (const float*)d_in[25];

  const int N = 40000, E = 640000;
  char* ws = (char*)d_ws;
  size_t off = 0;
  auto walloc = [&](size_t bytes) -> char* {
    char* p = ws + off;
    off = (off + bytes + 255) & ~(size_t)255;
    return p;
  };
  float*    h     = (float*)   walloc((size_t)N * 768 * 4);
  float*    feat1 = (float*)   walloc((size_t)N * 512 * 4);
  float*    rst1  = (float*)   walloc((size_t)N * 512 * 4);
  float*    feat2 = (float*)   walloc((size_t)N * 128 * 4);
  float*    rst2  = (float*)   walloc((size_t)N * 128 * 4);
  float*    el1   = (float*)   walloc((size_t)N * 4 * 4);
  float*    er1   = (float*)   walloc((size_t)N * 4 * 4);
  unsigned* emax1 = (unsigned*)walloc((size_t)N * 4 * 4);
  float*    den1  = (float*)   walloc((size_t)N * 4 * 4);
  float*    we1   = (float*)   walloc((size_t)E * 4 * 4);
  float*    el2   = (float*)   walloc((size_t)N * 4);
  float*    er2   = (float*)   walloc((size_t)N * 4);
  unsigned* emax2 = (unsigned*)walloc((size_t)N * 4);
  float*    den2  = (float*)   walloc((size_t)N * 4);
  float*    we2   = (float*)   walloc((size_t)E * 4);
  float*    hg    = (float*)   walloc(128 * 4);

  float* out = (float*)d_out;

  hipMemsetAsync(rst1, 0, (size_t)N * 512 * 4, stream);
  hipMemsetAsync(rst2, 0, (size_t)N * 128 * 4, stream);
  hipMemsetAsync(hg,   0, 128 * 4, stream);

  const int MT = (N + 255) / 256;

  // h = concat(interp(rand), interp(func), emb) @ proj_W + proj_b
  gemm_bf16_wmma<<<dim3(768 / 64, MT), 256, 0, stream>>>(
      nullptr, proj_W, proj_b, h, N, 2304, 768, rand_f, func_f, emb_f, 1);

  // ---- GAT layer 1 (H=4, Dh=128) ----
  gemm_bf16_wmma<<<dim3(512 / 64, MT), 256, 0, stream>>>(
      h, W1, nullptr, feat1, N, 768, 512, nullptr, nullptr, nullptr, 0);
  rowdot<<<(N * 4 * 32 + 255) / 256, 256, 0, stream>>>(
      feat1, al1, ar1, el1, er1, emax1, den1, N, 4, 128);
  edge_max<<<(E * 4 + 255) / 256, 256, 0, stream>>>(src, dst, el1, er1, emax1, E, 4);
  edge_exp<<<(E * 4 + 255) / 256, 256, 0, stream>>>(src, dst, el1, er1, emax1, we1, den1, E, 4);
  edge_agg<<<(unsigned)(((size_t)E * 4 * 32 + 255) / 256), 256, 0, stream>>>(
      src, dst, feat1, we1, den1, rst1, E, 4, 128);
  add_bias_rows<<<(N * 512 + 255) / 256, 256, 0, stream>>>(rst1, b1, N * 512, 512);

  // ---- GAT layer 2 (H=1, Dh=128) ----
  gemm_bf16_wmma<<<dim3(128 / 64, MT), 256, 0, stream>>>(
      rst1, W2, nullptr, feat2, N, 512, 128, nullptr, nullptr, nullptr, 0);
  rowdot<<<(N * 32 + 255) / 256, 256, 0, stream>>>(
      feat2, al2, ar2, el2, er2, emax2, den2, N, 1, 128);
  edge_max<<<(E + 255) / 256, 256, 0, stream>>>(src, dst, el2, er2, emax2, E, 1);
  edge_exp<<<(E + 255) / 256, 256, 0, stream>>>(src, dst, el2, er2, emax2, we2, den2, E, 1);
  edge_agg<<<(unsigned)(((size_t)E * 32 + 255) / 256), 256, 0, stream>>>(
      src, dst, feat2, we2, den2, rst2, E, 1, 128);

  // ---- heads ----
  node_head<<<N, 128, 0, stream>>>(rst2, b2, naW, nab, nmW1, nmb1, nmW2, nmb2,
                                   temp, out, hg);
  graph_head<<<1, 128, 0, stream>>>(hg, gmW1, gmb1, gmW2, gmb2, temp,
                                    out + (size_t)N * 2);
}
